// PPYOLOEHead_39462159516005
// MI455X (gfx1250) — matrix operand order repaired and checked by reading, not compiled, not run on hardware
//
#include <hip/hip_runtime.h>
#include <hip/hip_bf16.h>
#include <stdint.h>

typedef __attribute__((ext_vector_type(8)))  _Float16 v8h;
typedef __attribute__((ext_vector_type(16))) _Float16 v16h;
typedef __attribute__((ext_vector_type(8)))  float    v8f;
typedef __attribute__((ext_vector_type(4)))  float    v4f;
typedef __attribute__((ext_vector_type(4)))  unsigned u32x4;
typedef __attribute__((ext_vector_type(8)))  int      i32x8;
typedef __attribute__((ext_vector_type(4)))  int      i32x4;

#define BSZ   16
#define NCLS  80
#define REGC  68   // 4*(16+1)
#define OCPAD 80   // both heads padded to 80 rows (5 WMMA M-tiles)

// ---------------------------------------------------------------------------
// TDM: stage a 3D tile (dim0=W halves, dim1=R rows, dim2=nci channels) of f16
// activations global->LDS.  Descriptor per CDNA5 ISA 8.3-8.6.
// ---------------------------------------------------------------------------
__device__ inline void tdm_load_tile_f16(unsigned lds_byte_off,
                                         const _Float16* gptr,
                                         int W, int R, int nci, int L) {
    unsigned long long ga = (unsigned long long)(uintptr_t)gptr;
    u32x4 g0;
    g0[0] = 1u;                                   // count=1, user mode, no gather
    g0[1] = lds_byte_off;                         // lds_addr
    g0[2] = (unsigned)(ga & 0xffffffffu);         // global_addr[31:0]
    g0[3] = (unsigned)((ga >> 32) & 0x01ffffffu)  // global_addr[56:32]
          | (2u << 30);                           // type = 2 ("image")
    i32x8 g1;
    g1[0] = (int)(1u << 16);                      // data_size=1 -> 2 bytes/elem
    g1[1] = (int)(((unsigned)W & 0xffffu) << 16);       // tensor_dim0[15:0]
    g1[2] = (int)(((unsigned)R & 0xffffu) << 16);       // dim0 hi=0 | tensor_dim1[15:0]
    g1[3] = (int)(((unsigned)W & 0xffffu) << 16);       // dim1 hi=0 | tile_dim0=W
    g1[4] = (int)(((unsigned)R & 0xffffu)               // tile_dim1=R
          | (((unsigned)nci & 0xffffu) << 16));         // tile_dim2=nci
    g1[5] = W;                                          // tensor_dim0_stride[31:0]
    g1[6] = (int)(((unsigned)L & 0xffffu) << 16);       // d0s hi=0 | dim1_stride[15:0]
    g1[7] = (int)((unsigned)L >> 16);                   // tensor_dim1_stride[47:16]
    i32x4 g2;
    g2[0] = nci;                                  // tensor_dim2
    g2[1] = 0; g2[2] = 0; g2[3] = 0;              // no dim3 / no iterate
    i32x4 g3 = {0, 0, 0, 0};
#if __clang_major__ >= 23
    i32x8 g4 = {0, 0, 0, 0, 0, 0, 0, 0};
    __builtin_amdgcn_tensor_load_to_lds(g0, g1, g2, g3, g4, 0);
#else
    __builtin_amdgcn_tensor_load_to_lds(g0, g1, g2, g3, 0);
#endif
}

// ---------------------------------------------------------------------------
// 1) adaptive avg pool (H,W)->(1,1): one block per (b,c)
// ---------------------------------------------------------------------------
__global__ __launch_bounds__(256)
void avgpool_kernel(const float* __restrict__ feat, float* __restrict__ avg, int L) {
    int bc = blockIdx.x;
    const float* p = feat + (size_t)bc * L;
    float s = 0.f;
    for (int i = threadIdx.x; i < L; i += 256) s += p[i];
    __shared__ float red[256];
    red[threadIdx.x] = s;
    __syncthreads();
    for (int off = 128; off > 0; off >>= 1) {
        if (threadIdx.x < off) red[threadIdx.x] += red[threadIdx.x + off];
        __syncthreads();
    }
    if (threadIdx.x == 0) avg[bc] = red[0] / (float)L;
}

// ---------------------------------------------------------------------------
// 2) ESE fc (1x1 conv on pooled vec) + sigmoid
// ---------------------------------------------------------------------------
__global__ void fc_sigmoid_kernel(const float* __restrict__ avg,
                                  const float* __restrict__ w,
                                  const float* __restrict__ bias,
                                  float* __restrict__ att, int C) {
    int b = blockIdx.x, co = threadIdx.x;
    const float* a  = avg + (size_t)b * C;
    const float* wr = w + (size_t)co * C;
    float s = bias[co];
    for (int ci = 0; ci < C; ++ci) s += wr[ci] * a[ci];
    att[(size_t)b * C + co] = 1.f / (1.f + __expf(-s));
}

// ---------------------------------------------------------------------------
// 3) stem 1x1 conv as WMMA GEMM:  Y = swish(BN(W @ (feat*att)))  [+ feat]
//    block tile 128(M) x 64(N), 8 waves, wave tile 32x32 (2x2 WMMA)
// ---------------------------------------------------------------------------
__global__ __launch_bounds__(256)
void stem_wmma_kernel(const float* __restrict__ feat,   // [B,C,L]
                      const float* __restrict__ att,    // [B,C]
                      const float* __restrict__ Wm,     // [C,C]
                      const float* __restrict__ bn_g, const float* __restrict__ bn_b,
                      const float* __restrict__ bn_m, const float* __restrict__ bn_v,
                      _Float16* __restrict__ Z,         // [B,C,L] f16
                      int C, int L, int add_residual) {
    __shared__ __align__(16) _Float16 As[128 * 32];     // [m][k] k-contiguous
    __shared__ __align__(16) _Float16 Bs[64 * 32];      // [n][k] k-contiguous
    const int b  = blockIdx.z;
    const int m0 = blockIdx.y * 128;
    const int n0 = blockIdx.x * 64;
    const float* featB = feat + (size_t)b * C * L;
    const float* attB  = att + (size_t)b * C;
    const int tid  = threadIdx.x;
    const int lane = tid & 31, wave = tid >> 5;
    const int wm = wave >> 1, wn = wave & 1;

    v8f acc[2][2] = {};

    for (int k0 = 0; k0 < C; k0 += 32) {
        if (k0 + 32 < C)
            __builtin_prefetch(featB + (size_t)(k0 + 32) * L + n0, 0, 0);
        #pragma unroll
        for (int it = 0; it < 4; ++it) {           // A tile 128x32 f32->f16
            int idx = (it * 256 + tid) * 4;
            int r = idx >> 5, c = idx & 31;
            int gm = m0 + r;
            v4f wv = {0.f, 0.f, 0.f, 0.f};
            if (gm < C) wv = *(const v4f*)(Wm + (size_t)gm * C + k0 + c);
            _Float16* d = As + r * 32 + c;
            d[0] = (_Float16)wv[0]; d[1] = (_Float16)wv[1];
            d[2] = (_Float16)wv[2]; d[3] = (_Float16)wv[3];
        }
        #pragma unroll
        for (int it = 0; it < 2; ++it) {           // B tile transposed into LDS
            int idx = (it * 256 + tid) * 4;
            int kk = idx >> 6;
            int nn = idx & 63;
            int ci = k0 + kk;
            int l  = n0 + nn;
            float scale = attB[ci];
            v4f xv = {0.f, 0.f, 0.f, 0.f};
            if (l + 3 < L) {
                xv = *(const v4f*)(featB + (size_t)ci * L + l);
            } else {
                #pragma unroll
                for (int j = 0; j < 4; ++j)
                    if (l + j < L) xv[j] = featB[(size_t)ci * L + l + j];
            }
            Bs[(nn + 0) * 32 + kk] = (_Float16)(xv[0] * scale);
            Bs[(nn + 1) * 32 + kk] = (_Float16)(xv[1] * scale);
            Bs[(nn + 2) * 32 + kk] = (_Float16)(xv[2] * scale);
            Bs[(nn + 3) * 32 + kk] = (_Float16)(xv[3] * scale);
        }
        __syncthreads();

        const int fm   = lane & 15;
        const int half = lane >> 4;
        const int abk  = half ? 8 : 0;
        const int bbk  = half ? 8 : 0;
        v16h afrag[2], bfrag[2];
        #pragma unroll
        for (int i = 0; i < 2; ++i) {
            const _Float16* ap = As + (wm * 32 + i * 16 + fm) * 32 + abk;
            v8h alo = *(const v8h*)ap;
            v8h ahi = *(const v8h*)(ap + 16);
            afrag[i] = __builtin_shufflevector(alo, ahi,
                        0,1,2,3,4,5,6,7,8,9,10,11,12,13,14,15);
            const _Float16* bp = Bs + (wn * 32 + i * 16 + fm) * 32 + bbk;
            v8h blo = *(const v8h*)bp;
            v8h bhi = *(const v8h*)(bp + 16);
            bfrag[i] = __builtin_shufflevector(blo, bhi,
                        0,1,2,3,4,5,6,7,8,9,10,11,12,13,14,15);
        }
        #pragma unroll
        for (int i = 0; i < 2; ++i)
            #pragma unroll
            for (int j = 0; j < 2; ++j)
                acc[i][j] = __builtin_amdgcn_wmma_f32_16x16x32_f16(
                    false, afrag[i], false, bfrag[j], (short)0, acc[i][j], false, false);
        __syncthreads();
    }

    const int fm   = lane & 15;
    const int half = lane >> 4;
    #pragma unroll
    for (int i = 0; i < 2; ++i) {
        #pragma unroll
        for (int j = 0; j < 2; ++j) {
            int nG = n0 + wn * 32 + j * 16 + fm;
            if (nG >= L) continue;
            #pragma unroll
            for (int r = 0; r < 8; ++r) {
                int co = m0 + wm * 32 + i * 16 + half * 8 + r;
                if (co >= C) continue;
                float y = acc[i][j][r];
                float inv = __frsqrt_rn(bn_v[co] + 1e-5f);
                y = (y - bn_m[co]) * (bn_g[co] * inv) + bn_b[co];
                y = y / (1.f + __expf(-y));                 // swish
                if (add_residual) y += featB[(size_t)co * L + nG];
                Z[(size_t)b * C * L + (size_t)co * L + nG] = (_Float16)y;
            }
        }
    }
}

// ---------------------------------------------------------------------------
// 3b) pack 3x3 weights f32 [OC,C,3,3] -> f16 [9][OCPAD][C] (k-contiguous rows,
//     matching WMMA A-fragment layout; rows oc>=OC zero-padded)
// ---------------------------------------------------------------------------
__global__ void pack_weights_kernel(const float* __restrict__ Wt,
                                    _Float16* __restrict__ Wh, int C, int OC) {
    int idx = blockIdx.x * blockDim.x + threadIdx.x;
    int total = 9 * OCPAD * C;
    if (idx >= total) return;
    int tap = idx / (OCPAD * C);
    int rem = idx - tap * (OCPAD * C);
    int oc  = rem / C;
    int ci  = rem - oc * C;
    float v = (oc < OC) ? Wt[((size_t)oc * C + ci) * 9 + tap] : 0.f;
    Wh[idx] = (_Float16)v;
}

// ---------------------------------------------------------------------------
// 4) 3x3 pred head, TDM-staged implicit WMMA GEMM.
//    Block: 8 waves; wave w owns output columns n0+16w..+15, all 80 rows.
//    Per 32-channel chunk: wave0 TDM-loads [32 x R x W] f16 tile to LDS,
//    waits TENSORcnt, barrier; 9 taps x 5 M-tiles of WMMA per wave.
// ---------------------------------------------------------------------------
__global__ __launch_bounds__(256)
void head_wmma_kernel(const _Float16* __restrict__ Z,    // [B,C,L] f16
                      const _Float16* __restrict__ Wh,   // [9][OCPAD][C] f16
                      const float* __restrict__ bias,    // [OC]
                      float* __restrict__ out_reg,       // [B,REGC,L] (mode 1)
                      float* __restrict__ out_scores,    // scores base (mode 0)
                      int C, int L, int Hh, int Ww,
                      int OC, int Atot, int Aoff, int mode) {
    __shared__ __align__(16) _Float16 Bs[12800];   // 32ch x R x W, R*W <= 400
    const int b    = blockIdx.z;
    const int tid  = threadIdx.x;
    const int lane = tid & 31, wave = tid >> 5;
    const int fm   = lane & 15;
    const int half = lane >> 4;
    const int abk  = half ? 8 : 0;
    const int bbk  = half ? 16 : 0;
    const _Float16* Zb = Z + (size_t)b * C * L;

    const int nblk = blockIdx.x * 128;
    const int ys   = nblk / Ww;
    const int yeL  = min(nblk + 127, L - 1) / Ww;
    const int ys2  = max(ys - 1, 0);
    const int ye2  = min(yeL + 1, Hh - 1);
    const int R    = ye2 - ys2 + 1;
    const int rw   = R * Ww;

    const int l = nblk + wave * 16 + fm;           // this lane's output column
    const int y = l / Ww, x = l % Ww;

    v8f acc[5] = {};

    for (int k0 = 0; k0 < C; k0 += 32) {
        __syncthreads();                           // prior chunk fully consumed
        if (wave == 0) {
            tdm_load_tile_f16(0u, Zb + (size_t)k0 * L + (size_t)ys2 * Ww,
                              Ww, R, 32, L);
            __builtin_amdgcn_s_wait_tensorcnt(0);
        }
        __syncthreads();                           // tile visible to all waves

        for (int tap = 0; tap < 9; ++tap) {
            const int dy = tap / 3 - 1, dx = tap % 3 - 1;
            const int iy = y + dy, ix = x + dx;
            const bool vcol = (l < L) && iy >= 0 && iy < Hh && ix >= 0 && ix < Ww;
            const int base = (iy - ys2) * Ww + ix;
            v16h bfrag;
            #pragma unroll
            for (int i = 0; i < 16; ++i)
                bfrag[i] = vcol ? Bs[(abk + (i & 7) + ((i >> 3) << 4)) * rw + base] : (_Float16)0.f;
            #pragma unroll
            for (int mt = 0; mt < 5; ++mt) {
                const _Float16* ap = Wh + ((size_t)tap * OCPAD + mt * 16 + fm) * C
                                        + k0 + abk;
                v8h alo = *(const v8h*)ap;
                v8h ahi = *(const v8h*)(ap + 16);
                v16h afrag = __builtin_shufflevector(alo, ahi,
                              0,1,2,3,4,5,6,7,8,9,10,11,12,13,14,15);
                acc[mt] = __builtin_amdgcn_wmma_f32_16x16x32_f16(
                    false, afrag, false, bfrag, (short)0, acc[mt], false, false);
            }
        }
    }

    #pragma unroll
    for (int mt = 0; mt < 5; ++mt) {
        #pragma unroll
        for (int r = 0; r < 8; ++r) {
            int oc = mt * 16 + half * 8 + r;
            if (oc >= OC || l >= L) continue;
            float v = acc[mt][r] + bias[oc];
            if (mode == 0) {
                out_scores[((size_t)b * NCLS + oc) * Atot + Aoff + l] =
                    1.f / (1.f + __expf(-v));
            } else {
                out_reg[((size_t)b * REGC + oc) * L + l] = v;
            }
        }
    }
}

// ---------------------------------------------------------------------------
// 5) DFL softmax-expectation + distance2bbox decode
// ---------------------------------------------------------------------------
__global__ void dfl_decode_kernel(const float* __restrict__ reg,   // [B,REGC,L]
                                  float* __restrict__ bbox,        // [B,Atot,4]
                                  int L, int Ww, int Atot, int Aoff, float stride) {
    int idx = blockIdx.x * blockDim.x + threadIdx.x;
    if (idx >= BSZ * L) return;
    int b = idx / L, l = idx % L;
    const float* rb = reg + (size_t)b * REGC * L + l;
    float dist[4];
    #pragma unroll
    for (int s = 0; s < 4; ++s) {
        float mx = -1e30f;
        for (int m = 0; m < 17; ++m)
            mx = fmaxf(mx, rb[(size_t)(s * 17 + m) * L]);
        float den = 0.f, num = 0.f;
        for (int m = 0; m < 17; ++m) {
            float e = __expf(rb[(size_t)(s * 17 + m) * L] - mx);
            den += e;
            num += e * (float)m;
        }
        dist[s] = num / den;
    }
    float xx = (float)(l % Ww) + 0.5f;
    float yy = (float)(l / Ww) + 0.5f;
    float* o = bbox + ((size_t)b * Atot + Aoff + l) * 4;
    o[0] = (xx - dist[0]) * stride;
    o[1] = (yy - dist[1]) * stride;
    o[2] = (xx + dist[2]) * stride;
    o[3] = (yy + dist[3]) * stride;
}

// ---------------------------------------------------------------------------
// launcher
// ---------------------------------------------------------------------------
extern "C" void kernel_launch(void* const* d_in, const int* in_sizes, int n_in,
                              void* d_out, int out_size, void* d_ws, size_t ws_size,
                              hipStream_t stream) {
    const int Cs[3] = {192, 384, 768};
    const int Hs[3] = {80, 40, 20};
    const float strd[3] = {8.f, 16.f, 32.f};
    const int Atot = 80 * 80 + 40 * 40 + 20 * 20;       // 8400
    const int Aoffs[3] = {0, 6400, 8000};

    float* out    = (float*)d_out;
    float* bbox   = out;                                 // [B, Atot, 4]
    float* scores = out + (size_t)BSZ * Atot * 4;        // [B, 80, Atot]

    // per-level params in sorted-key (JAX dict flatten) order:
    // 0 fc_cls_b  1 fc_cls_w  2 fc_reg_b  3 fc_reg_w  4 pred_cls_b 5 pred_cls_w
    // 6 pred_reg_b 7 pred_reg_w 8 stem_cls_be 9 stem_cls_g 10 stem_cls_m
    // 11 stem_cls_v 12 stem_cls_w 13 stem_reg_be 14 stem_reg_g 15 stem_reg_m
    // 16 stem_reg_v 17 stem_reg_w
    auto P = [&](int lev, int k) -> const float* {
        return (const float*)d_in[3 + 18 * lev + k];
    };

    char* ws = (char*)d_ws;
    size_t off = 0;
    auto alloc = [&](size_t bytes) -> char* {
        char* p = ws + off;
        off = (off + bytes + 255) & ~(size_t)255;
        return p;
    };

    for (int lev = 0; lev < 3; ++lev) {
        const int C = Cs[lev], Hh = Hs[lev], Ww = Hs[lev], L = Hh * Ww;
        const float* feat = (const float*)d_in[lev];

        float*     avg    = (float*)alloc((size_t)BSZ * C * sizeof(float));
        float*     wcls   = (float*)alloc((size_t)BSZ * C * sizeof(float));
        float*     wreg   = (float*)alloc((size_t)BSZ * C * sizeof(float));
        _Float16*  Zcls   = (_Float16*)alloc((size_t)BSZ * C * L * sizeof(_Float16));
        _Float16*  Zreg   = (_Float16*)alloc((size_t)BSZ * C * L * sizeof(_Float16));
        _Float16*  WhCls  = (_Float16*)alloc((size_t)9 * OCPAD * C * sizeof(_Float16));
        _Float16*  WhReg  = (_Float16*)alloc((size_t)9 * OCPAD * C * sizeof(_Float16));
        float*     regout = (float*)alloc((size_t)BSZ * REGC * L * sizeof(float));

        // 1) pooled features
        avgpool_kernel<<<BSZ * C, 256, 0, stream>>>(feat, avg, L);

        // 2) ESE attention weights
        fc_sigmoid_kernel<<<BSZ, C, 0, stream>>>(avg, P(lev, 1), P(lev, 0), wcls, C);
        fc_sigmoid_kernel<<<BSZ, C, 0, stream>>>(avg, P(lev, 3), P(lev, 2), wreg, C);

        // 2b) pack 3x3 head weights to f16 [tap][oc][ci]
        int wtot = 9 * OCPAD * C;
        pack_weights_kernel<<<(wtot + 255) / 256, 256, 0, stream>>>(
            P(lev, 5), WhCls, C, NCLS);
        pack_weights_kernel<<<(wtot + 255) / 256, 256, 0, stream>>>(
            P(lev, 7), WhReg, C, REGC);

        // 3) stems (WMMA GEMM, fused BN+swish, +residual for cls)
        dim3 sgrid((L + 63) / 64, (C + 127) / 128, BSZ);
        stem_wmma_kernel<<<sgrid, 256, 0, stream>>>(
            feat, wcls, P(lev, 12), P(lev, 9), P(lev, 8), P(lev, 10), P(lev, 11),
            Zcls, C, L, /*residual=*/1);
        stem_wmma_kernel<<<sgrid, 256, 0, stream>>>(
            feat, wreg, P(lev, 17), P(lev, 14), P(lev, 13), P(lev, 15), P(lev, 16),
            Zreg, C, L, /*residual=*/0);

        // 4) 3x3 pred heads (TDM-staged implicit WMMA GEMM)
        dim3 hgrid((L + 127) / 128, 1, BSZ);
        head_wmma_kernel<<<hgrid, 256, 0, stream>>>(
            Zcls, WhCls, P(lev, 4), nullptr, scores,
            C, L, Hh, Ww, NCLS, Atot, Aoffs[lev], /*mode=*/0);
        head_wmma_kernel<<<hgrid, 256, 0, stream>>>(
            Zreg, WhReg, P(lev, 6), regout, nullptr,
            C, L, Hh, Ww, REGC, Atot, Aoffs[lev], /*mode=*/1);

        // 5) DFL + bbox decode
        int nthr = BSZ * L;
        dfl_decode_kernel<<<(nthr + 255) / 256, 256, 0, stream>>>(
            regout, bbox, L, Ww, Atot, Aoffs[lev], strd[lev]);
    }
}
